// Attention_69441031241911
// MI455X (gfx1250) — compile-verified, matrix operand order checked by reading, not run on hardware
//
#include <hip/hip_runtime.h>
#include <hip/hip_bf16.h>

#define HID   4096
#define QKV_N 6144
#define B_    4
#define Q_    1024
#define H_    32
#define HKV_  8
#define D_    128
#define NTOK  (B_ * Q_)

typedef __attribute__((ext_vector_type(16))) __bf16        v16bf;
typedef __attribute__((ext_vector_type(8)))  float         v8f;
typedef __attribute__((ext_vector_type(4)))  unsigned int  u32x4;
typedef __attribute__((ext_vector_type(8)))  int           i32x8;
typedef __attribute__((ext_vector_type(4)))  int           i32x4;

union Frag16 { v16bf v; unsigned short u[16]; u32x4 q[2]; };
union F8     { v8f   v; float f[8]; };

__device__ __forceinline__ unsigned short f2bf(float f) {
  unsigned int u = __float_as_uint(f);
  u += 0x7FFFu + ((u >> 16) & 1u);          // round-to-nearest-even
  return (unsigned short)(u >> 16);
}

__device__ __forceinline__ v8f wmma_bf16(v16bf a, v16bf b, v8f c) {
  return __builtin_amdgcn_wmma_f32_16x16x32_bf16(false, a, false, b, (short)0, c,
                                                 false, false);
}

// --------------------------------------------------------------- TDM issue
// 2D bf16 tile (tile_w elems x tile_h rows) from global -> LDS, with LDS row
// padding 16B per 64B row (80B stride). D# bitfields per CDNA5 ISA §8.
__device__ __forceinline__ void tdm_load_2d(
    const void* gaddr, unsigned lds_off, int rowlen_elems, int nrows,
    int tile_w, int tile_h) {
  const unsigned long long ga = (unsigned long long)gaddr;
  u32x4 g0;
  g0.x = 1u;                                    // count=1, user load
  g0.y = lds_off;                               // LDS byte address
  g0.z = (unsigned)ga;                          // global_addr[31:0]
  g0.w = (unsigned)(ga >> 32) | (2u << 30);     // global_addr msbs | type=2
  i32x8 g1;
  g1[0] = (1 << 16)        // data_size = 2 bytes
        | (1 << 20)        // pad_enable
        | (3 << 22)        // pad_interval: 16 DWORDs (one 64B tile row)
        | (3 << 25);       // pad_amount: 4 DWORDs -> 80B LDS row stride
  g1[1] = (rowlen_elems & 0xFFFF) << 16;                          // dim0[15:0]
  g1[2] = (int)(((unsigned)rowlen_elems >> 16) |
                ((unsigned)(nrows & 0xFFFF) << 16));              // dim0 hi, dim1 lo
  g1[3] = (int)(((unsigned)nrows >> 16) |
                ((unsigned)(tile_w & 0xFFFF) << 16));             // dim1 hi, tile_dim0
  g1[4] = tile_h & 0xFFFF;                                        // tile_dim1, tile_dim2=0
  g1[5] = rowlen_elems;                                           // dim0_stride[31:0]
  g1[6] = 0;                                                      // stride hi, dim1_stride
  g1[7] = 0;
  const i32x4 z4 = {0, 0, 0, 0};
  const i32x8 z8 = {0, 0, 0, 0, 0, 0, 0, 0};
  __builtin_amdgcn_tensor_load_to_lds(g0, g1, z4, z4, z8, 0);
}

// ------------------------------------------------------------- activations cvt
__global__ __launch_bounds__(256) void cvt_f32_bf16(
    const float* __restrict__ src, unsigned short* __restrict__ dst, long n) {
  long i = ((long)blockIdx.x * blockDim.x + threadIdx.x) * 8;
  const long stride = (long)gridDim.x * blockDim.x * 8;
  for (; i < n; i += stride) {
    float4 a = *(const float4*)(src + i);
    float4 b = *(const float4*)(src + i + 4);
    u32x4 o;
    o.x = (unsigned)f2bf(a.x) | ((unsigned)f2bf(a.y) << 16);
    o.y = (unsigned)f2bf(a.z) | ((unsigned)f2bf(a.w) << 16);
    o.z = (unsigned)f2bf(b.x) | ((unsigned)f2bf(b.y) << 16);
    o.w = (unsigned)f2bf(b.z) | ((unsigned)f2bf(b.w) << 16);
    *(u32x4*)(dst + i) = o;
  }
}

// ---------------------------------------------------------------- weight cvt
// src[K][N] f32 -> dst[N][K] bf16 (32x32 LDS tile transpose)
__global__ __launch_bounds__(256) void cvt_transpose_bf16(
    const float* __restrict__ src, unsigned short* __restrict__ dst,
    int K, int N) {
  __shared__ unsigned short tile[32][34];
  const int k0 = blockIdx.y * 32, n0 = blockIdx.x * 32;
  const int tx = threadIdx.x & 31, ty = threadIdx.x >> 5;   // 32 x 8
#pragma unroll
  for (int i = ty; i < 32; i += 8)
    tile[i][tx] = f2bf(src[(long)(k0 + i) * N + n0 + tx]);
  __syncthreads();
#pragma unroll
  for (int i = ty; i < 32; i += 8)
    dst[(long)(n0 + i) * K + k0 + tx] = tile[tx][i];
}

// ------------------------------------------------------------------ GEMM
// C[M,N](f32) = A[M,K](bf16 row-major) x W^T[N,K](bf16)
// 128x128 block tile, 8 waves (2x4), wave tile 64x32. Tiles are DMA'd by the
// Tensor Data Mover into double-buffered LDS (TENSORcnt-tracked), overlapping
// the WMMA work on the front buffer.
__global__ __launch_bounds__(256) void gemm_wmma(
    const unsigned short* __restrict__ Ab, const unsigned short* __restrict__ WbT,
    float* __restrict__ C, int M, int N, int K) {
  __shared__ unsigned short lA[2][128][40];   // [buf][m][k], 80B row stride
  __shared__ unsigned short lB[2][128][40];   // [buf][n][k]

  const int tid   = threadIdx.x;
  const int lane  = tid & 31;
  const int wave  = tid >> 5;
  const int half  = lane >> 4;
  const int ln    = lane & 15;
  const int base8 = half * 8;
  const int wm    = wave >> 2;             // 0..1
  const int wn    = wave & 3;              // 0..3
  const long row0 = (long)blockIdx.y * 128;
  const long col0 = (long)blockIdx.x * 128;

  const unsigned ldsA[2] = {(unsigned)(unsigned long long)&lA[0][0][0],
                            (unsigned)(unsigned long long)&lA[1][0][0]};
  const unsigned ldsB[2] = {(unsigned)(unsigned long long)&lB[0][0][0],
                            (unsigned)(unsigned long long)&lB[1][0][0]};

  F8 acc[4][2];
#pragma unroll
  for (int i = 0; i < 4; ++i)
#pragma unroll
    for (int j = 0; j < 2; ++j)
#pragma unroll
      for (int r = 0; r < 8; ++r) acc[i][j].f[r] = 0.f;

  const int NK = K >> 5;                   // 32-wide K steps
  if (tid < 32) {                          // wave 0 drives the TDM
    tdm_load_2d(Ab + row0 * K, ldsA[0], K, M, 32, 128);
    tdm_load_2d(WbT + col0 * K, ldsB[0], K, N, 32, 128);
  }

  for (int it = 0; it < NK; ++it) {
    const int cur = it & 1;
    __syncthreads();                       // back buffer free to overwrite
    if (tid < 32) {
      if (it + 1 < NK) {
        const int k1 = (it + 1) << 5;
        tdm_load_2d(Ab + row0 * K + k1, ldsA[cur ^ 1], K, M, 32, 128);
        tdm_load_2d(WbT + col0 * K + k1, ldsB[cur ^ 1], K, N, 32, 128);
        __builtin_amdgcn_s_wait_tensorcnt(2);   // front pair done (in-order)
      } else {
        __builtin_amdgcn_s_wait_tensorcnt(0);
      }
    }
    __syncthreads();                       // front buffer visible to all waves

    Frag16 aF[4], bF[2];
#pragma unroll
    for (int i = 0; i < 4; ++i) {
      int m = wm * 64 + i * 16 + ln;
      aF[i].q[0] = *(const u32x4*)&lA[cur][m][base8];
      aF[i].q[1] = *(const u32x4*)&lA[cur][m][16 + base8];
    }
#pragma unroll
    for (int j = 0; j < 2; ++j) {
      int n = wn * 32 + j * 16 + ln;
      bF[j].q[0] = *(const u32x4*)&lB[cur][n][base8];
      bF[j].q[1] = *(const u32x4*)&lB[cur][n][16 + base8];
    }
#pragma unroll
    for (int i = 0; i < 4; ++i)
#pragma unroll
      for (int j = 0; j < 2; ++j)
        acc[i][j].v = wmma_bf16(aF[i].v, bF[j].v, acc[i][j].v);
  }

#pragma unroll
  for (int i = 0; i < 4; ++i)
#pragma unroll
    for (int j = 0; j < 2; ++j)
#pragma unroll
      for (int r = 0; r < 8; ++r) {
        long row = row0 + wm * 64 + i * 16 + r + base8;
        long col = col0 + wn * 32 + j * 16 + ln;
        C[row * N + col] = acc[i][j].f[r];
      }
}

// -------------------------------------------------- fused RoPE + KV scatter
// qb: bf16 [tok][H][D], roped q pre-scaled by D^-0.5 (feeds score A-frags)
// kc: bf16 [slot][HKV][D] (row-major K rows feed score B-frags directly)
// vt: bf16 [block][HKV][D][16] (transposed V feeds PV B-frags directly)
__global__ __launch_bounds__(256) void rope_scatter(
    const float* __restrict__ qkv, unsigned short* __restrict__ qb,
    unsigned short* __restrict__ kc, unsigned short* __restrict__ vt,
    const int* __restrict__ positions, const int* __restrict__ slot_mapping) {
  const int tok = blockIdx.x;
  const int tid = threadIdx.x;
  const float pos = (float)positions[tok];
  const int slot = slot_mapping[tok];
  const float scale = 0.0883883476483184f;                // D^-0.5
  const float* base = qkv + (long)tok * QKV_N;

  for (int idx = tid; idx < H_ * 64; idx += 256) {        // rope q -> bf16 (scaled)
    int h = idx >> 6, i = idx & 63;
    float inv = __expf(-(float)i * 0.14391156f);          // ln(10000)/64
    float s, c, ang = pos * inv;
    __sincosf(ang, &s, &c);
    const float* p = base + h * D_;
    float x1 = p[i], x2 = p[i + 64];
    unsigned short* qd = qb + ((long)tok * H_ + h) * D_;
    qd[i]      = f2bf((x1 * c - x2 * s) * scale);
    qd[i + 64] = f2bf((x1 * s + x2 * c) * scale);
  }
  for (int idx = tid; idx < HKV_ * 64; idx += 256) {      // rope k -> bf16 cache
    int h = idx >> 6, i = idx & 63;
    float inv = __expf(-(float)i * 0.14391156f);
    float s, c, ang = pos * inv;
    __sincosf(ang, &s, &c);
    const float* p = base + HID + h * D_;
    float x1 = p[i], x2 = p[i + 64];
    unsigned short* kd = kc + ((long)slot * HKV_ + h) * D_;
    kd[i]      = f2bf(x1 * c - x2 * s);
    kd[i + 64] = f2bf(x1 * s + x2 * c);
  }
  for (int idx = tid; idx < HKV_ * D_; idx += 256) {      // v -> transposed cache
    int h = idx >> 7, d = idx & 127;
    vt[((((long)(slot >> 4)) * HKV_ + h) * D_ + d) * 16 + (slot & 15)] =
        f2bf(base[HID + HKV_ * D_ + idx]);
  }
}

// ------------------------------------------- flash attention (WMMA, wave32)
// one wave per (b, h, 16-row q tile); 32-key steps; online softmax; all
// Q/K/V fragments are direct 16B global loads from bf16 buffers; output
// written directly in bf16 for the w_o GEMM.
__global__ __launch_bounds__(32) void flash_attn(
    const unsigned short* __restrict__ qb, const unsigned short* __restrict__ kc,
    const unsigned short* __restrict__ vt, const int* __restrict__ bt_all,
    unsigned short* __restrict__ attn) {
  const int lane  = threadIdx.x & 31;
  const int half  = lane >> 4;
  const int ln    = lane & 15;
  const int base8 = half * 8;
  const int q0  = blockIdx.x * 16;
  const int h   = blockIdx.y;
  const int b   = blockIdx.z;
  const int kvh = h >> 2;                                  // H/HKV = 4
  const int* bt = bt_all + b * (Q_ / 16);

  __shared__ unsigned short pLds[16][34];                  // P transpose buffer

  Frag16 aQ[4];
  {
    const unsigned short* qrow =
        qb + ((long)(b * Q_ + q0 + ln) * H_ + h) * D_;
#pragma unroll
    for (int c = 0; c < 4; ++c) {
      aQ[c].q[0] = *(const u32x4*)(qrow + c * 32 + base8);
      aQ[c].q[1] = *(const u32x4*)(qrow + c * 32 + 16 + base8);
    }
  }

  float m_r[8], l_r[8];
  F8 O[8];
#pragma unroll
  for (int r = 0; r < 8; ++r) { m_r[r] = -3.0e38f; l_r[r] = 0.f; }
#pragma unroll
  for (int t = 0; t < 8; ++t)
#pragma unroll
    for (int r = 0; r < 8; ++r) O[t].f[r] = 0.f;

  const int kmax = q0 + 16;                                // visible keys (mult of 16)
  const int nsteps = (kmax + 31) >> 5;
  for (int st = 0; st < nsteps; ++st) {
    const int s0 = st * 32;

    // ---- scores: two 16x16 tiles over this 32-key step
    F8 sc[2];
#pragma unroll
    for (int t2 = 0; t2 < 2; ++t2) {
#pragma unroll
      for (int r = 0; r < 8; ++r) sc[t2].f[r] = 0.f;
      int key = s0 + t2 * 16 + ln;
      if (key >= kmax) key -= 16;          // clamp to prev valid tile (masked)
      const unsigned short* krow =
          kc + ((long)(bt[key >> 4] * 16 + (key & 15)) * HKV_ + kvh) * D_;
#pragma unroll
      for (int c = 0; c < 4; ++c) {
        Frag16 bK;
        bK.q[0] = *(const u32x4*)(krow + c * 32 + base8);
        bK.q[1] = *(const u32x4*)(krow + c * 32 + 16 + base8);
        sc[t2].v = wmma_bf16(aQ[c].v, bK.v, sc[t2].v);
      }
    }

    // ---- causal mask + online softmax (row = r + 8*half of C layout)
    float p0[8], p1[8];
#pragma unroll
    for (int r = 0; r < 8; ++r) {
      const int row = q0 + r + base8;
      float s0v = ((s0 + ln) <= row)      ? sc[0].f[r] : -3.0e38f;
      float s1v = ((s0 + 16 + ln) <= row) ? sc[1].f[r] : -3.0e38f;
      float mloc = fmaxf(s0v, s1v);
#pragma unroll
      for (int mk = 1; mk <= 8; mk <<= 1)
        mloc = fmaxf(mloc, __shfl_xor(mloc, mk, 32));
      const float mnew  = fmaxf(m_r[r], mloc);
      const float alpha = __expf(m_r[r] - mnew);
      const float e0 = __expf(s0v - mnew);
      const float e1 = __expf(s1v - mnew);
      float rsum = e0 + e1;
#pragma unroll
      for (int mk = 1; mk <= 8; mk <<= 1) rsum += __shfl_xor(rsum, mk, 32);
      l_r[r] = l_r[r] * alpha + rsum;
      m_r[r] = mnew;
#pragma unroll
      for (int t = 0; t < 8; ++t) O[t].f[r] *= alpha;
      p0[r] = e0;
      p1[r] = e1;
    }

    // ---- P: C-layout -> A-layout via LDS
#pragma unroll
    for (int r = 0; r < 8; ++r) {
      pLds[r + base8][ln]      = f2bf(p0[r]);
      pLds[r + base8][16 + ln] = f2bf(p1[r]);
    }
    __syncthreads();
    Frag16 aP;
#pragma unroll
    for (int e = 0; e < 8; ++e) {
      aP.u[e]     = pLds[ln][base8 + e];
      aP.u[8 + e] = pLds[ln][16 + base8 + e];
    }
    __syncthreads();

    // ---- O += P @ V  (K = 32 keys; 8-aligned runs are all-valid or all-masked)
    int r0 = s0 + base8;      if (r0 >= kmax) r0 = kmax - 8;
    int r1 = s0 + 16 + base8; if (r1 >= kmax) r1 = kmax - 8;
    const unsigned short* v0 =
        vt + ((long)bt[r0 >> 4] * HKV_ + kvh) * D_ * 16 + (r0 & 15);
    const unsigned short* v1 =
        vt + ((long)bt[r1 >> 4] * HKV_ + kvh) * D_ * 16 + (r1 & 15);
#pragma unroll
    for (int t = 0; t < 8; ++t) {
      Frag16 bV;
      const int dcol = t * 16 + ln;
      bV.q[0] = *(const u32x4*)(v0 + (long)dcol * 16);
      bV.q[1] = *(const u32x4*)(v1 + (long)dcol * 16);
      O[t].v = wmma_bf16(aP.v, bV.v, O[t].v);
    }
  }

  // ---- normalize + write bf16 (A-operand of the w_o GEMM)
#pragma unroll
  for (int r = 0; r < 8; ++r) {
    const float inv = 1.0f / l_r[r];
    unsigned short* orow =
        attn + (long)(b * Q_ + q0 + r + base8) * HID + h * D_;
#pragma unroll
    for (int t = 0; t < 8; ++t) orow[t * 16 + ln] = f2bf(O[t].f[r] * inv);
  }
}

// -------------------------------------------------------------------- launch
extern "C" void kernel_launch(void* const* d_in, const int* in_sizes, int n_in,
                              void* d_out, int out_size, void* d_ws, size_t ws_size,
                              hipStream_t stream) {
  const float* hs         = (const float*)d_in[0];
  const float* wqkv       = (const float*)d_in[1];
  const float* wo         = (const float*)d_in[2];
  const int* positions    = (const int*)d_in[5];
  const int* slot_mapping = (const int*)d_in[6];
  const int* block_tables = (const int*)d_in[7];
  float* out = (float*)d_out;

  char* p = (char*)d_ws;
  unsigned short* wqkvT = (unsigned short*)p; p += (size_t)HID * QKV_N * 2;  // [QKV_N][HID]
  unsigned short* woT   = (unsigned short*)p; p += (size_t)HID * HID * 2;    // [HID][HID]
  unsigned short* hs_bf = (unsigned short*)p; p += (size_t)NTOK * HID * 2;   // [NTOK][HID]
  float* qkv  = (float*)p;                    p += (size_t)NTOK * QKV_N * 4;
  unsigned short* qb = (unsigned short*)p;    p += (size_t)NTOK * H_ * D_ * 2;
  unsigned short* kc = (unsigned short*)p;    p += (size_t)NTOK * HKV_ * D_ * 2;
  unsigned short* vt = (unsigned short*)p;    p += (size_t)NTOK * HKV_ * D_ * 2;
  unsigned short* attn = (unsigned short*)p;  // [NTOK][HID]

  cvt_f32_bf16<<<2048, 256, 0, stream>>>(hs, hs_bf, (long)NTOK * HID);
  {
    dim3 gt(QKV_N / 32, HID / 32);
    cvt_transpose_bf16<<<gt, 256, 0, stream>>>(wqkv, wqkvT, HID, QKV_N);
  }
  {
    dim3 gt(HID / 32, HID / 32);
    cvt_transpose_bf16<<<gt, 256, 0, stream>>>(wo, woT, HID, HID);
  }

  dim3 g1(QKV_N / 128, NTOK / 128);
  gemm_wmma<<<g1, 256, 0, stream>>>(hs_bf, wqkvT, qkv, NTOK, QKV_N, HID);

  rope_scatter<<<NTOK, 256, 0, stream>>>(qkv, qb, kc, vt, positions, slot_mapping);

  dim3 g2(Q_ / 16, H_, B_);
  flash_attn<<<g2, 32, 0, stream>>>(qb, kc, vt, block_tables, attn);

  dim3 g3(HID / 128, NTOK / 128);
  gemm_wmma<<<g3, 256, 0, stream>>>(attn, woT, out, NTOK, HID, HID);
}
